// PhysicsSparseMoE_12927851561752
// MI455X (gfx1250) — matrix operand order repaired and checked by reading, not compiled, not run on hardware
//
#include <hip/hip_runtime.h>
#include <hip/hip_bf16.h>
#include <math.h>

#define DIMC  768
#define HIDC  3072
#define NEXP  17
#define TBINS 16
#define NB    8
#define NTOK  1024
#define BNTOT (NB * NTOK)
#define KEYM  3
#define LDSP  40   // LDS row pitch (halves) for 32-wide K tiles (bank-conflict pad)

typedef __attribute__((ext_vector_type(16))) _Float16 v16h;
typedef __attribute__((ext_vector_type(8)))  _Float16 v8h;
typedef __attribute__((ext_vector_type(4)))  _Float16 v4h;
typedef __attribute__((ext_vector_type(8)))  float    v8f;

__device__ __forceinline__ float gelu_exact(float v) {
    return 0.5f * v * (1.0f + erff(v * 0.70710678118654752440f));
}

// ---------------------------------------------------------------------------
// Kernel 1: time-aware summary -> x_video[b, 2*DIMC]
// One block per video, 256 threads. All reductions deterministic.
// ---------------------------------------------------------------------------
__global__ __launch_bounds__(256) void summary_kernel(
    const float* __restrict__ x, const int* __restrict__ time_ids,
    float* __restrict__ x_video)
{
    __shared__ int   s_tid[NTOK];
    __shared__ float s_h[TBINS][DIMC];
    __shared__ int   s_cnt[TBINS];
    __shared__ float s_ga[DIMC];
    __shared__ float s_sg[TBINS];
    __shared__ float s_sd[TBINS];
    __shared__ float s_w[KEYM];
    __shared__ int   s_idx[KEYM];
    __shared__ int   s_nvalid;

    const int b   = blockIdx.x;
    const int tid = threadIdx.x;
    const float* xb = x + (size_t)b * NTOK * DIMC;

    for (int i = tid; i < TBINS * DIMC; i += 256) ((float*)s_h)[i] = 0.0f;
    if (tid < TBINS) s_cnt[tid] = 0;
    __syncthreads();

    for (int n = tid; n < NTOK; n += 256) {
        int t = time_ids[b * NTOK + n];
        s_tid[n] = t;
        atomicAdd(&s_cnt[t], 1);
    }
    __syncthreads();

    // per-bin sums then means; each channel owned by exactly one thread
    for (int c = tid; c < DIMC; c += 256) {
        for (int n = 0; n < NTOK; ++n)
            s_h[s_tid[n]][c] += xb[(size_t)n * DIMC + c];
        for (int t = 0; t < TBINS; ++t)
            s_h[t][c] = s_h[t][c] / ((float)s_cnt[t] + 1e-6f);
    }
    __syncthreads();

    if (tid == 0) {
        int nv = 0;
        for (int t = 0; t < TBINS; ++t) nv += (s_cnt[t] > 0) ? 1 : 0;
        s_nvalid = nv;
    }
    __syncthreads();

    for (int c = tid; c < DIMC; c += 256) {
        float sum = 0.0f;
        for (int t = 0; t < TBINS; ++t)
            if (s_cnt[t] > 0) sum += s_h[t][c];
        s_ga[c] = sum / fmaxf((float)s_nvalid, 1.0f);
    }
    __syncthreads();

    if (tid < TBINS) {
        const int t  = tid;
        const int tp = (t + TBINS - 1) % TBINS;
        const bool vcur  = s_cnt[t] > 0;
        const bool vprev = (t != 0) && (s_cnt[tp] > 0);
        float sg = 0.0f, sd = 0.0f;
        for (int c = 0; c < DIMC; ++c) {
            float h = s_h[t][c];
            sg += fabsf(h - s_ga[c]);
            sd += fabsf(h - s_h[tp][c]);
        }
        s_sg[t] = sg;
        s_sd[t] = (vcur && vprev) ? sd : 0.0f;
    }
    __syncthreads();

    if (tid == 0) {
        float s[TBINS];
        for (int t = 0; t < TBINS; ++t)
            s[t] = (s_cnt[t] > 0) ? (0.5f * s_sg[t] + 0.5f * s_sd[t]) : -1e9f;
        bool used[TBINS] = {};
        float tv[KEYM]; int tix[KEYM];
        for (int j = 0; j < KEYM; ++j) {
            float best = -INFINITY; int bi = 0;
            for (int t = 0; t < TBINS; ++t)
                if (!used[t] && s[t] > best) { best = s[t]; bi = t; }
            used[bi] = true; tv[j] = best; tix[j] = bi;
        }
        float m = tv[0];
        float esum = 0.0f, ev[KEYM];
        for (int j = 0; j < KEYM; ++j) { ev[j] = expf(tv[j] - m); esum += ev[j]; }
        for (int j = 0; j < KEYM; ++j) { s_w[j] = ev[j] / esum; s_idx[j] = tix[j]; }
    }
    __syncthreads();

    float* xv = x_video + (size_t)b * (2 * DIMC);
    for (int c = tid; c < DIMC; c += 256) {
        xv[c] = s_ga[c];
        xv[DIMC + c] = s_w[0] * s_h[s_idx[0]][c]
                     + s_w[1] * s_h[s_idx[1]][c]
                     + s_w[2] * s_h[s_idx[2]][c];
    }
}

// ---------------------------------------------------------------------------
// Kernel 2: video gate -> top-1 expert per video + balance loss
// ---------------------------------------------------------------------------
__global__ __launch_bounds__(256) void router_kernel(
    const float* __restrict__ x_video, const float* __restrict__ gate_w,
    int* __restrict__ expert_idx, float* __restrict__ loss_out)
{
    __shared__ float s_log[NB][NEXP];
    __shared__ float s_sc[NB][NEXP];
    __shared__ int   s_exp[NB];
    const int tid = threadIdx.x;

    if (tid < NB * NEXP) {
        int b = tid / NEXP, e = tid % NEXP;
        const float* xv = x_video + (size_t)b * 2 * DIMC;
        const float* gw = gate_w + (size_t)e * 2 * DIMC;
        float acc = 0.0f;
        for (int k = 0; k < 2 * DIMC; ++k) acc += xv[k] * gw[k];
        s_log[b][e] = acc;
    }
    __syncthreads();

    if (tid < NB) {
        int b = tid;
        float m = s_log[b][0];
        for (int e = 1; e < NEXP; ++e) m = fmaxf(m, s_log[b][e]);
        float sum = 0.0f;
        for (int e = 0; e < NEXP; ++e) { float v = expf(s_log[b][e] - m); s_sc[b][e] = v; sum += v; }
        float inv = 1.0f / sum;
        int best = 0; float bv = -INFINITY;
        for (int e = 0; e < NEXP; ++e) {
            s_sc[b][e] *= inv;
            if (s_sc[b][e] > bv) { bv = s_sc[b][e]; best = e; }
        }
        s_exp[b] = best;
        expert_idx[b] = best;
    }
    __syncthreads();

    if (tid == 0) {
        int cnt[NEXP] = {};
        for (int b = 0; b < NB; ++b) cnt[s_exp[b]]++;
        float loss = 0.0f;
        for (int e = 0; e < NEXP; ++e) {
            float mean = 0.0f;
            for (int b = 0; b < NB; ++b) mean += s_sc[b][e];
            mean *= (1.0f / (float)NB);
            float f = ((float)cnt[e] * (float)NTOK) / ((float)BNTOT + 1e-6f);
            loss += f * mean;
        }
        loss_out[0] = (float)NEXP * loss;
    }
}

// ---------------------------------------------------------------------------
// WMMA GEMM core: C[128x128] tile, 8 waves (4 along M x 2 along N),
// wave tile 32x64 = 2x4 accumulators, K staged in double-buffered LDS as f16.
// Staging loops are compile-time unrolled so all global loads of a tile are
// issued back-to-back (batched s_wait_loadcnt drain) and overlap with WMMA.
// MODE 0: H = gelu(x @ fc1_w^T + b1)         -> half  [NTOK, HIDC] per video
// MODE 1: agg = H @ fc2_w^T + b2             -> float [NTOK, DIMC] per video
// MODE 2: Hf = gelu(agg @ fus_w1[:, :768]^T + fus_w1[:, 768+e] + fus_b1) -> half
// MODE 3: out = Hf @ fus_w2^T + fus_b2 + agg -> float (d_out)
// ---------------------------------------------------------------------------
struct GemmArgs {
    const float* x;
    const float* fc1_w; const float* fc1_b;
    const float* fc2_w; const float* fc2_b;
    const float* fus_w1; const float* fus_b1;
    const float* fus_w2; const float* fus_b2;
    const int*   expert;
    _Float16*    H;     // [BNTOT*HIDC] halves; reused as Hf [BNTOT*DIMC]
    float*       agg;   // [BNTOT*DIMC]
    float*       out;   // [BNTOT*DIMC]
};

template <int MODE>
__global__ __launch_bounds__(256) void moe_gemm(GemmArgs g)
{
    constexpr int KTOT = (MODE == 1) ? HIDC : DIMC;
    constexpr int LDW  = (MODE == 2) ? (DIMC + NEXP) : KTOT;
    constexpr int KS   = KTOT / 32;

    __shared__ _Float16 As[2][128 * LDSP];
    __shared__ _Float16 Bs[2][128 * LDSP];

    const int v    = blockIdx.z;
    const int e    = g.expert[v];
    const int tid  = threadIdx.x;
    const int lane = tid & 31;
    const int wave = tid >> 5;
    const int wm   = wave & 3;
    const int wn   = wave >> 2;
    const int m0   = blockIdx.x * 128;   // token row within this video
    const int n0   = blockIdx.y * 128;   // output column tile

    const float*    Af = nullptr;
    const _Float16* Ah = nullptr;
    const float*    W;
    const float*    bias;
    if constexpr (MODE == 0) {
        Af = g.x + (size_t)v * NTOK * DIMC;
        W  = g.fc1_w + (size_t)e * HIDC * DIMC;
        bias = g.fc1_b + (size_t)e * HIDC;
    } else if constexpr (MODE == 1) {
        Ah = g.H + (size_t)v * NTOK * HIDC;
        W  = g.fc2_w + (size_t)e * DIMC * HIDC;
        bias = g.fc2_b + (size_t)e * DIMC;
    } else if constexpr (MODE == 2) {
        Af = g.agg + (size_t)v * NTOK * DIMC;
        W  = g.fus_w1;
        bias = g.fus_b1;
    } else {
        Ah = g.H + (size_t)v * NTOK * DIMC;
        W  = g.fus_w2;
        bias = g.fus_b2;
    }

    // ---- staging: constant trip counts, loads batched before converts ----
    auto loadA = [&](int buf, int k0) {
        if constexpr (MODE == 0 || MODE == 2) {
            const float* src = Af + (size_t)m0 * KTOT + k0;
            float4 q[4];
            #pragma unroll
            for (int j = 0; j < 4; ++j) {
                int i = tid + j * 256;                 // 1024 float4 per tile
                int r = i >> 3, c = (i & 7) << 2;
                q[j] = *(const float4*)(src + (size_t)r * KTOT + c);
            }
            #pragma unroll
            for (int j = 0; j < 4; ++j) {
                int i = tid + j * 256;
                int r = i >> 3, c = (i & 7) << 2;
                v4h h; h[0] = (_Float16)q[j].x; h[1] = (_Float16)q[j].y;
                       h[2] = (_Float16)q[j].z; h[3] = (_Float16)q[j].w;
                *(v4h*)&As[buf][r * LDSP + c] = h;
            }
            if ((tid & 63) == 0) __builtin_prefetch(src + 32, 0, 1);
        } else {
            const _Float16* src = Ah + (size_t)m0 * KTOT + k0;
            v8h q[2];
            #pragma unroll
            for (int j = 0; j < 2; ++j) {
                int i = tid + j * 256;                 // 512 v8h per tile
                int r = i >> 2, c = (i & 3) << 3;
                q[j] = *(const v8h*)(src + (size_t)r * KTOT + c);
            }
            #pragma unroll
            for (int j = 0; j < 2; ++j) {
                int i = tid + j * 256;
                int r = i >> 2, c = (i & 3) << 3;
                *(v8h*)&As[buf][r * LDSP + c] = q[j];
            }
            if ((tid & 63) == 0) __builtin_prefetch(src + 32, 0, 1);
        }
    };

    auto loadB = [&](int buf, int k0) {
        const float* src = W + (size_t)n0 * LDW + k0;
        if constexpr ((LDW & 3) == 0) {
            float4 q[4];
            #pragma unroll
            for (int j = 0; j < 4; ++j) {
                int i = tid + j * 256;
                int r = i >> 3, c = (i & 7) << 2;
                q[j] = *(const float4*)(src + (size_t)r * LDW + c);
            }
            #pragma unroll
            for (int j = 0; j < 4; ++j) {
                int i = tid + j * 256;
                int r = i >> 3, c = (i & 7) << 2;
                v4h h; h[0] = (_Float16)q[j].x; h[1] = (_Float16)q[j].y;
                       h[2] = (_Float16)q[j].z; h[3] = (_Float16)q[j].w;
                *(v4h*)&Bs[buf][r * LDSP + c] = h;
            }
        } else {
            float q[16];
            #pragma unroll
            for (int j = 0; j < 16; ++j) {
                int i = tid + j * 256;                 // 4096 scalars per tile
                int r = i >> 5, c = i & 31;
                q[j] = src[(size_t)r * LDW + c];
            }
            #pragma unroll
            for (int j = 0; j < 16; ++j) {
                int i = tid + j * 256;
                int r = i >> 5, c = i & 31;
                Bs[buf][r * LDSP + c] = (_Float16)q[j];
            }
        }
        if ((tid & 63) == 0) __builtin_prefetch(src + 32, 0, 1);
    };

    // A fragment (16x32 f16): lanes 0-15 -> M=lane, K in {0..7,16..23};
    // lanes 16-31 -> M=lane-16, K in {8..15,24..31}.
    auto a_frag = [&](const _Float16* S, int rb) -> v16h {
        int r  = rb + (lane & 15);
        int ko = (lane & 16) ? 8 : 0;
        const _Float16* p = S + r * LDSP + ko;
        v8h lo = *(const v8h*)p;
        v8h hi = *(const v8h*)(p + 16);
        return __builtin_shufflevector(lo, hi, 0, 1, 2, 3, 4, 5, 6, 7,
                                               8, 9, 10, 11, 12, 13, 14, 15);
    };
    // B fragment (32x16 f16, W stored N-major [N,K]): lanes 0-15 -> N=lane,
    // K in {0..7,16..23}; lanes 16-31 -> N=lane-16, K in {8..15,24..31}.
    auto b_frag = [&](const _Float16* S, int cb) -> v16h {
        int c  = cb + (lane & 15);
        int kb = (lane & 16) ? 8 : 0;
        const _Float16* p = S + c * LDSP + kb;
        v8h lo = *(const v8h*)p;
        v8h hi = *(const v8h*)(p + 16);
        return __builtin_shufflevector(lo, hi, 0, 1, 2, 3, 4, 5, 6, 7,
                                               8, 9, 10, 11, 12, 13, 14, 15);
    };

    v8f acc[2][4] = {};

    loadA(0, 0);
    loadB(0, 0);
    __syncthreads();

    for (int ks = 0; ks < KS; ++ks) {
        const int buf = ks & 1;
        if (ks + 1 < KS) {
            loadA(buf ^ 1, (ks + 1) * 32);
            loadB(buf ^ 1, (ks + 1) * 32);
        }
        v16h a0 = a_frag(As[buf], wm * 32);
        v16h a1 = a_frag(As[buf], wm * 32 + 16);
        v16h b0 = b_frag(Bs[buf], wn * 64);
        v16h b1 = b_frag(Bs[buf], wn * 64 + 16);
        v16h b2 = b_frag(Bs[buf], wn * 64 + 32);
        v16h b3 = b_frag(Bs[buf], wn * 64 + 48);

        acc[0][0] = __builtin_amdgcn_wmma_f32_16x16x32_f16(false, a0, false, b0, (short)0, acc[0][0], false, false);
        acc[0][1] = __builtin_amdgcn_wmma_f32_16x16x32_f16(false, a0, false, b1, (short)0, acc[0][1], false, false);
        acc[0][2] = __builtin_amdgcn_wmma_f32_16x16x32_f16(false, a0, false, b2, (short)0, acc[0][2], false, false);
        acc[0][3] = __builtin_amdgcn_wmma_f32_16x16x32_f16(false, a0, false, b3, (short)0, acc[0][3], false, false);
        acc[1][0] = __builtin_amdgcn_wmma_f32_16x16x32_f16(false, a1, false, b0, (short)0, acc[1][0], false, false);
        acc[1][1] = __builtin_amdgcn_wmma_f32_16x16x32_f16(false, a1, false, b1, (short)0, acc[1][1], false, false);
        acc[1][2] = __builtin_amdgcn_wmma_f32_16x16x32_f16(false, a1, false, b2, (short)0, acc[1][2], false, false);
        acc[1][3] = __builtin_amdgcn_wmma_f32_16x16x32_f16(false, a1, false, b3, (short)0, acc[1][3], false, false);
        __syncthreads();
    }

    // Epilogue. C/D layout: element r of v8f -> M = r + 8*(lane>=16), N = lane&15.
    const int lm = (lane & 16) ? 8 : 0;
    const int ln = lane & 15;
    const size_t tokbase = (size_t)v * NTOK + (size_t)m0;

    #pragma unroll
    for (int ms = 0; ms < 2; ++ms) {
        #pragma unroll
        for (int ns = 0; ns < 4; ++ns) {
            const int col = n0 + wn * 64 + ns * 16 + ln;
            float bi = bias[col];
            if constexpr (MODE == 2) bi += g.fus_w1[(size_t)col * LDW + DIMC + e];
            #pragma unroll
            for (int r = 0; r < 8; ++r) {
                const size_t tok = tokbase + (size_t)(wm * 32 + ms * 16 + lm + r);
                float val = acc[ms][ns][r] + bi;
                if constexpr (MODE == 0) {
                    g.H[tok * HIDC + col] = (_Float16)gelu_exact(val);
                } else if constexpr (MODE == 1) {
                    g.agg[tok * DIMC + col] = val;
                } else if constexpr (MODE == 2) {
                    g.H[tok * DIMC + col] = (_Float16)gelu_exact(val);
                } else {
                    g.out[tok * DIMC + col] = val + g.agg[tok * DIMC + col];
                }
            }
        }
    }
}

// ---------------------------------------------------------------------------
extern "C" void kernel_launch(void* const* d_in, const int* in_sizes, int n_in,
                              void* d_out, int out_size, void* d_ws, size_t ws_size,
                              hipStream_t stream)
{
    (void)in_sizes; (void)n_in; (void)out_size; (void)ws_size;
    const float* x              = (const float*)d_in[0];
    const int*   time_ids       = (const int*)d_in[1];
    // d_in[2] = num_time_bins (== TBINS, compile-time constant here)
    const float* gate_w         = (const float*)d_in[3];
    const float* token_router_w = (const float*)d_in[4]; (void)token_router_w; // drops out for TOPK=1
    const float* fc1_w          = (const float*)d_in[5];
    const float* fc1_b          = (const float*)d_in[6];
    const float* fc2_w          = (const float*)d_in[7];
    const float* fc2_b          = (const float*)d_in[8];
    const float* fus_w1         = (const float*)d_in[9];
    const float* fus_b1         = (const float*)d_in[10];
    const float* fus_w2         = (const float*)d_in[11];
    const float* fus_b2         = (const float*)d_in[12];

    char* ws = (char*)d_ws;
    float*    x_video = (float*)ws;                         // 49152 B
    int*      expert  = (int*)(ws + 49152);                 // 32 B
    float*    agg     = (float*)(ws + 65536);               // 25165824 B
    _Float16* H       = (_Float16*)(ws + 65536 + 25165824); // 50331648 B

    float* out  = (float*)d_out;
    float* loss = out + (size_t)BNTOT * DIMC;

    summary_kernel<<<NB, 256, 0, stream>>>(x, time_ids, x_video);
    router_kernel<<<1, 256, 0, stream>>>(x_video, gate_w, expert, loss);

    GemmArgs g;
    g.x = x; g.fc1_w = fc1_w; g.fc1_b = fc1_b;
    g.fc2_w = fc2_w; g.fc2_b = fc2_b;
    g.fus_w1 = fus_w1; g.fus_b1 = fus_b1;
    g.fus_w2 = fus_w2; g.fus_b2 = fus_b2;
    g.expert = expert; g.H = H; g.agg = agg; g.out = out;

    dim3 blk(256);
    moe_gemm<0><<<dim3(NTOK / 128, HIDC / 128, NB), blk, 0, stream>>>(g); // fc1 + gelu
    moe_gemm<1><<<dim3(NTOK / 128, DIMC / 128, NB), blk, 0, stream>>>(g); // fc2 -> agg
    moe_gemm<2><<<dim3(NTOK / 128, DIMC / 128, NB), blk, 0, stream>>>(g); // fusion1 + gelu
    moe_gemm<3><<<dim3(NTOK / 128, DIMC / 128, NB), blk, 0, stream>>>(g); // fusion2 + residual
}
